// CRNN_8246337208597
// MI455X (gfx1250) — matrix operand order repaired and checked by reading, not compiled, not run on hardware
//
#include <hip/hip_runtime.h>
#include <math.h>

// Problem sizes (fixed by the reference)
#define TT 4096
#define II 1024
#define HH 1024

typedef __attribute__((ext_vector_type(16))) _Float16 v16h;
typedef __attribute__((ext_vector_type(8)))  _Float16 v8h;
typedef __attribute__((ext_vector_type(4)))  _Float16 v4h;
typedef __attribute__((ext_vector_type(8)))  float    v8f;

// ---------------------------------------------------------------------------
// Phase 0: one-shot f32 -> f16 conversion (x and W_ih), bandwidth-bound.
// ---------------------------------------------------------------------------
__global__ __launch_bounds__(256) void crnn_cvt_f16(
    const float* __restrict__ s, _Float16* __restrict__ d, int n4) {
  const int i = blockIdx.x * 256 + threadIdx.x;
  if (i < n4) {
    const float4 v = reinterpret_cast<const float4*>(s)[i];
    v4h o;
    o[0] = (_Float16)v.x; o[1] = (_Float16)v.y;
    o[2] = (_Float16)v.z; o[3] = (_Float16)v.w;
    reinterpret_cast<v4h*>(d)[i] = o;
  }
}

// ---------------------------------------------------------------------------
// Phase 1: x_proj = x @ W_ih^T + (b_ih + b_hh)   (complex, f16 WMMA, f32 acc)
// Block = 128 threads (4 waves). Block tile = 64 (T) x 64 (H). K step 32.
// Each wave computes a 16x64 strip: A fragment loaded once, reused across
// 4 B column-tiles -> 16 v_wmma per K-step per wave.
// ---------------------------------------------------------------------------

// A fragment (16x32, contiguous k order, same as B): lane m<16 holds row m,
// K 0..15; lane m+16 holds K 16..31.
__device__ __forceinline__ v16h load_a_f16(const _Float16* __restrict__ base) {
  const v8h lo = *reinterpret_cast<const v8h*>(base);       // K .. K+7
  const v8h hi = *reinterpret_cast<const v8h*>(base + 8);   // K+8 .. K+15
  return __builtin_shufflevector(lo, hi, 0, 1, 2, 3, 4, 5, 6, 7,
                                 8, 9, 10, 11, 12, 13, 14, 15);
}

__global__ __launch_bounds__(128) void crnn_xproj_gemm(
    const _Float16* __restrict__ xhr, const _Float16* __restrict__ xhi,
    const _Float16* __restrict__ whr, const _Float16* __restrict__ whi,
    const float* __restrict__ bihr, const float* __restrict__ bihi,
    const float* __restrict__ bhhr, const float* __restrict__ bhhi,
    float* __restrict__ xproj /* (T*H) interleaved complex f32 */) {
  // B tile: 64 W-rows x 32 K halves, row stride 48 halves (96B) keeps the
  // 32B v16h fragment reads aligned.
  __shared__ __attribute__((aligned(32))) _Float16 Bsr[64][48];
  __shared__ __attribute__((aligned(32))) _Float16 Bsi[64][48];

  const int tid   = threadIdx.x;
  const int wave  = tid >> 5;
  const int lane  = tid & 31;
  const int mbase = blockIdx.x * 64 + wave * 16;  // time-tile base for this wave
  const int nbase = blockIdx.y * 64;              // hidden-tile base (4 sub-tiles)
  const int mrow  = mbase + (lane & 15);
  const int khalf = (lane >> 4) * 16;             // A K sub-offset (contiguous k order, 0 or 16)

  const _Float16* __restrict__ xr_row = xhr + (size_t)mrow * II + khalf;
  const _Float16* __restrict__ xi_row = xhi + (size_t)mrow * II + khalf;

  // LDS fill mapping: 128 threads cover 64 rows x 32 halves as 2 x 32B copies
  const int frow = tid >> 1;            // 0..63
  const int fkk  = (tid & 1) * 16;      // 0 or 16 halves

  v8f Prr[4] = {}, Pii[4] = {}, Pri[4] = {}, Pir[4] = {};

  for (int k0 = 0; k0 < II; k0 += 32) {
    // --- stage W_ih rows (B = W^T) into LDS: one 32B load+store per matrix ---
    {
      const size_t g = (size_t)(nbase + frow) * II + k0 + fkk;
      *reinterpret_cast<v16h*>(&Bsr[frow][fkk]) =
          *reinterpret_cast<const v16h*>(&whr[g]);
      *reinterpret_cast<v16h*>(&Bsi[frow][fkk]) =
          *reinterpret_cast<const v16h*>(&whi[g]);
    }
    __syncthreads();

    if (k0 + 32 < II) {  // gfx1250 global_prefetch of next A tile (near caches)
      __builtin_prefetch(xr_row + k0 + 32, 0, 3);
      __builtin_prefetch(xi_row + k0 + 32, 0, 3);
    }

    const v16h ar = load_a_f16(xr_row + k0);
    const v16h ai = load_a_f16(xi_row + k0);

#pragma unroll
    for (int nt = 0; nt < 4; ++nt) {
      // B fragment: lane n holds W[nt*16+n][k0 + (lane>=16?16:0) .. +15]
      const v16h br = *reinterpret_cast<const v16h*>(
          &Bsr[nt * 16 + (lane & 15)][(lane >> 4) * 16]);
      const v16h bi = *reinterpret_cast<const v16h*>(
          &Bsi[nt * 16 + (lane & 15)][(lane >> 4) * 16]);
      Prr[nt] = __builtin_amdgcn_wmma_f32_16x16x32_f16(false, ar, false, br,
                                                       (short)0, Prr[nt], false, false);
      Pii[nt] = __builtin_amdgcn_wmma_f32_16x16x32_f16(false, ai, false, bi,
                                                       (short)0, Pii[nt], false, false);
      Pri[nt] = __builtin_amdgcn_wmma_f32_16x16x32_f16(false, ar, false, bi,
                                                       (short)0, Pri[nt], false, false);
      Pir[nt] = __builtin_amdgcn_wmma_f32_16x16x32_f16(false, ai, false, br,
                                                       (short)0, Pir[nt], false, false);
    }
    __syncthreads();
  }

  // Epilogue: C/D layout = VGPR r -> M=r (lanes 0-15) / M=r+8 (lanes 16-31)
#pragma unroll
  for (int nt = 0; nt < 4; ++nt) {
    const int col = nbase + nt * 16 + (lane & 15);
    const float bRv = bihr[col] + bhhr[col];
    const float bIv = bihi[col] + bhhi[col];
#pragma unroll
    for (int r = 0; r < 8; ++r) {
      const int t  = mbase + r + ((lane >> 4) << 3);
      const float re = Prr[nt][r] - Pii[nt][r] + bRv;
      const float im = Pri[nt][r] + Pir[nt][r] + bIv;
      reinterpret_cast<float2*>(xproj)[(size_t)t * HH + col] = make_float2(re, im);
    }
  }
}

// ---------------------------------------------------------------------------
// Phase 2: persistent recurrent scan. 128 blocks x 256 threads.
// Each block pins 8 rows of W_hh (complex f32, 64KB) in LDS; h ping-pongs in
// L2 (8KB); one atomic grid barrier per timestep.
// ---------------------------------------------------------------------------
#define NBLK 128
#define CPB  8   // channels (W_hh rows) per block

__global__ __launch_bounds__(256) void crnn_scan(
    const float* __restrict__ whr, const float* __restrict__ whi,
    const float* __restrict__ xproj,
    float* __restrict__ hA, float* __restrict__ hB,
    float* __restrict__ out, int* __restrict__ bar) {
  __shared__ float Wf[CPB * HH * 2];  // [(c*HH + k)*2 + {re,im}] = 64KB

  const int tid   = threadIdx.x;
  const int wave  = tid >> 5;
  const int lane  = tid & 31;
  const int cbase = blockIdx.x * CPB;

  // Pin this block's W_hh rows in LDS (f32, interleaved re/im).
  for (int i = tid; i < CPB * HH; i += 256) {
    const int c = i >> 10;
    const int k = i & (HH - 1);
    Wf[2 * i + 0] = whr[(size_t)(cbase + c) * HH + k];
    Wf[2 * i + 1] = whi[(size_t)(cbase + c) * HH + k];
  }
  __syncthreads();

  const int c_local = wave;        // 8 waves -> 8 channels
  const int kb      = lane * 32;   // 32 K values per lane
  const float2* __restrict__ wp2 =
      reinterpret_cast<const float2*>(&Wf[(c_local * HH + kb) * 2]);

  for (int t = 0; t < TT; ++t) {
    const float* hcur = (t & 1) ? hB : hA;
    float*       hnxt = (t & 1) ? hA : hB;
    const float2* __restrict__ hp = reinterpret_cast<const float2*>(hcur);

    float ar = 0.f, ai = 0.f;
#pragma unroll 8
    for (int k = 0; k < 32; ++k) {
      const float2 hv = hp[kb + k];
      const float2 wv = wp2[k];
      ar = fmaf(hv.x, wv.x, ar);
      ar = fmaf(-hv.y, wv.y, ar);
      ai = fmaf(hv.x, wv.y, ai);
      ai = fmaf(hv.y, wv.x, ai);
    }
    // wave32 reduction
#pragma unroll
    for (int off = 1; off < 32; off <<= 1) {
      ar += __shfl_xor(ar, off, 32);
      ai += __shfl_xor(ai, off, 32);
    }

    if (lane == 0) {
      const int c = cbase + c_local;
      const float2 xp = reinterpret_cast<const float2*>(xproj)[(size_t)t * HH + c];
      const float zr = ar + xp.x;
      const float zi = ai + xp.y;
      // stable complex tanh: (sinh 2a + i sin 2b) / (cosh 2a + cos 2b)
      const float ta = 2.f * zr, tb = 2.f * zi;
      float hr, hi;
      if (fabsf(ta) > 40.f) {
        hr = copysignf(1.f, zr);
        hi = 0.f;
      } else {
        const float den = coshf(ta) + cosf(tb);
        hr = sinhf(ta) / den;
        hi = sinf(tb) / den;
      }
      reinterpret_cast<float2*>(hnxt)[c] = make_float2(hr, hi);
      reinterpret_cast<float2*>(out)[(size_t)t * HH + c] = make_float2(hr, hi);
    }
    __threadfence();

    // ---- grid-wide barrier (counter + generation), one per timestep ----
    __syncthreads();
    if (tid == 0) {
      const int g = __atomic_load_n(&bar[1], __ATOMIC_ACQUIRE);
      if (atomicAdd(&bar[0], 1) == NBLK - 1) {
        atomicExch(&bar[0], 0);
        __threadfence();
        __atomic_fetch_add(&bar[1], 1, __ATOMIC_RELEASE);
      } else {
        while (__atomic_load_n(&bar[1], __ATOMIC_ACQUIRE) == g) {
          __builtin_amdgcn_s_sleep(1);
        }
      }
    }
    __syncthreads();
  }
}

// Deterministic per-launch init: zero h buffers + barrier words.
__global__ void crnn_init(float* hA, float* hB, int* bar) {
  const int i = threadIdx.x;
  for (int j = i; j < 2 * HH; j += 256) {
    hA[j] = 0.f;
    hB[j] = 0.f;
  }
  if (i < 2) bar[i] = 0;
}

extern "C" void kernel_launch(void* const* d_in, const int* in_sizes, int n_in,
                              void* d_out, int out_size, void* d_ws, size_t ws_size,
                              hipStream_t stream) {
  (void)in_sizes; (void)n_in; (void)out_size; (void)ws_size;
  const float* xr   = (const float*)d_in[0];
  const float* xi   = (const float*)d_in[1];
  const float* Wihr = (const float*)d_in[2];
  const float* Wihi = (const float*)d_in[3];
  const float* Whhr = (const float*)d_in[4];
  const float* Whhi = (const float*)d_in[5];
  const float* bihr = (const float*)d_in[6];
  const float* bihi = (const float*)d_in[7];
  const float* bhhr = (const float*)d_in[8];
  const float* bhhi = (const float*)d_in[9];
  float* out = (float*)d_out;  // (1,T,H) complex64 -> interleaved f32 pairs

  // Workspace layout (bytes)
  char* ws = (char*)d_ws;
  float*     xproj = (float*)ws;                                     // 32MB
  _Float16*  xh_r  = (_Float16*)(ws + (size_t)TT * HH * 2 * 4);      //  8MB
  _Float16*  xh_i  = xh_r + (size_t)TT * II;                         //  8MB
  _Float16*  wh_r  = xh_i + (size_t)TT * II;                         //  2MB
  _Float16*  wh_i  = wh_r + (size_t)HH * II;                         //  2MB
  float*     hA    = (float*)(wh_i + (size_t)HH * II);               //  8KB
  float*     hB    = hA + 2 * HH;                                    //  8KB
  int*       bar   = (int*)(hB + 2 * HH);

  crnn_init<<<1, 256, 0, stream>>>(hA, hB, bar);

  // Phase 0: one-shot f16 conversions
  const int nx4 = (TT * II) / 4;
  const int nw4 = (HH * II) / 4;
  crnn_cvt_f16<<<nx4 / 256, 256, 0, stream>>>(xr, xh_r, nx4);
  crnn_cvt_f16<<<nx4 / 256, 256, 0, stream>>>(xi, xh_i, nx4);
  crnn_cvt_f16<<<nw4 / 256, 256, 0, stream>>>(Wihr, wh_r, nw4);
  crnn_cvt_f16<<<nw4 / 256, 256, 0, stream>>>(Wihi, wh_i, nw4);

  // Phase 1: WMMA input projection
  dim3 g1(TT / 64, HH / 64);  // 64 x 16 blocks
  crnn_xproj_gemm<<<g1, 128, 0, stream>>>(xh_r, xh_i, wh_r, wh_i,
                                          bihr, bihi, bhhr, bhhi, xproj);

  // Phase 2: persistent scan
  crnn_scan<<<NBLK, 256, 0, stream>>>(Whhr, Whhi, xproj, hA, hB, out, bar);
}
